// UserManager_NAML_21818433864463
// MI455X (gfx1250) — matrix operand order repaired and checked by reading, not compiled, not run on hardware
//
#include <hip/hip_runtime.h>
#include <hip/hip_bf16.h>

typedef __attribute__((ext_vector_type(16))) __bf16 v16bf;
typedef __attribute__((ext_vector_type(8)))  __bf16 v8bf;
typedef __attribute__((ext_vector_type(8)))  float  v8f;

#define BB   128
#define HH   100
#define CC   32
#define KCH  4
#define DD   400
#define EE   200
#define CWF  100
#define NTOT (BB*CC)      /* 4096 */

/* ---- kernel 1 LDS geometry (bf16 elements) ---- */
#define SA_ROWS 112
#define SA_STR  424       /* >=416, 16B-aligned stride, bank-friendly */
#define SBT_COLS 208
#define SBT_STR 40        /* >=32, 16B-aligned stride */

/* ---- kernel 2 LDS geometry ---- */
#define K2_NPB  16        /* candidates (n) per block     */
#define K2_ROWS 64        /* (n,k) rows per block         */
#define SA2_STR 136       /* >=128, 16B-aligned stride    */
#define SB2_STR 136

/* CDNA5-native tanh: gfx1250 adds V_TANH_F32 (TRANS op). Prefer the clang
   builtin (backend handles TRANS hazards); fall back to inline asm with a
   trailing v_nop to satisfy the 1-op TRANS RAW spacing rule (ISA ch5 7.4). */
__device__ inline float tanh_fast(float x) {
#if __has_builtin(__builtin_amdgcn_tanhf)
  return __builtin_amdgcn_tanhf(x);
#else
  float y;
  asm volatile("v_tanh_f32 %0, %1\n\tv_nop" : "=v"(y) : "v"(x));
  return y;
#endif
}

/* A-fragment (16x32 bf16, MxK): lane holds row = lane&15;
   elements 0..15 -> K = 32*kchunk + ((lane<16)?0:16) + 0..15,
   where kcol = 32*kchunk + ((lane<16)?0:8).  Two contiguous 16B LDS reads. */
__device__ inline v16bf load_a_frag(const __bf16* base, int stride, int row, int kcol) {
  const __bf16* p = base + row * stride + kcol + (kcol & 8);
  v8bf lo = *(const v8bf*)(p);
  v8bf hi = *(const v8bf*)(p + 8);
  v16bf r;
#pragma unroll
  for (int i = 0; i < 8; ++i) { r[i] = lo[i]; r[i + 8] = hi[i]; }
  return r;
}

/* B-fragment (32x16 bf16, KxN) from K-contiguous ("transposed") LDS tile:
   lane holds col = lane&15; elements 0..15 -> K = krow+0..15,
   where krow = ((lane<16)?0:16).  One contiguous 32B LDS read. */
__device__ inline v16bf load_b_frag(const __bf16* base, int stride, int col, int krow) {
  const __bf16* p = base + col * stride + krow;
  v8bf lo = *(const v8bf*)(p);
  v8bf hi = *(const v8bf*)(p + 8);
  v16bf r;
#pragma unroll
  for (int i = 0; i < 8; ++i) { r[i] = lo[i]; r[i + 8] = hi[i]; }
  return r;
}

/* =====================================================================
 * Kernel 1: one block per (b,k).
 * All loops touching acc[] are force-unrolled (static VGPR indexing).
 * ===================================================================== */
__global__ void __launch_bounds__(256, 1) attn_kernel(
    const float* __restrict__ his,  const float* __restrict__ W_att,
    const float* __restrict__ b_att, const float* __restrict__ q_att,
    const float* __restrict__ W_lin, const float* __restrict__ b_lin,
    float* __restrict__ out, float* __restrict__ g1ws)
{
  extern __shared__ char smem[];
  __bf16* sA  = (__bf16*)smem;                                   /* [112][424] his (bf16) */
  __bf16* sBt = (__bf16*)(smem + SA_ROWS * SA_STR * 2);          /* [208][40] W chunk^T   */
  float* logits = (float*)(smem + SA_ROWS * SA_STR * 2 + SBT_COLS * SBT_STR * 2);
  float* aw  = logits + SA_ROWS;
  float* u_s = aw + SA_ROWS;                                     /* [400] */

  const int blk = blockIdx.x;
  const int b = blk / KCH, k = blk % KCH;
  const int tid = threadIdx.x;
  const int wave = tid >> 5, lane = tid & 31;
  const int l15 = lane & 15;
  const int khalf = (lane < 16) ? 0 : 8;
  const int kb2   = (lane < 16) ? 0 : 16;

  /* stage his[b,:,k,:] -> bf16, zero-padded to 112x424 */
  const float* hisb = his + (size_t)b * HH * (KCH * DD) + (size_t)k * DD;
  for (int idx = tid; idx < SA_ROWS * SA_STR; idx += 256) {
    int h = idx / SA_STR, d = idx - h * SA_STR;
    float v = (h < HH && d < DD) ? hisb[(size_t)h * (KCH * DD) + d] : 0.0f;
    sA[idx] = (__bf16)v;
  }

  v8f acc[13];
#pragma unroll
  for (int n = 0; n < 13; ++n)
#pragma unroll
    for (int r = 0; r < 8; ++r) acc[n][r] = 0.0f;

  const float* Wk = W_att + (size_t)k * DD * EE;

  for (int kc = 0; kc < 13; ++kc) {
    __syncthreads();                       /* sA ready / previous compute done */
    /* stage sBt[e][r] = W_att[k, kc*32+r, e] (bf16, zero-padded) */
    for (int idx = tid; idx < 32 * SBT_COLS; idx += 256) {
      int r = idx / SBT_COLS, e = idx - r * SBT_COLS;
      int d = kc * 32 + r;
      float v = (d < DD && e < EE) ? Wk[(size_t)d * EE + e] : 0.0f;
      sBt[e * SBT_STR + r] = (__bf16)v;
    }
    __syncthreads();
    if (wave < 7) {
      v16bf afrag = load_a_frag(sA, SA_STR, wave * 16 + l15, kc * 32 + khalf);
#pragma unroll
      for (int n = 0; n < 13; ++n) {
        v16bf bfrag = load_b_frag(sBt, SBT_STR, n * 16 + l15, kb2);
        acc[n] = __builtin_amdgcn_wmma_f32_16x16x32_bf16(
            false, afrag, false, bfrag, (short)0, acc[n], false, false);
      }
    }
  }
  __syncthreads();

  /* epilogue: tanh(+bias), weight by q_att, reduce over E -> logits[h] */
  if (wave < 7) {
    float pd[8];
#pragma unroll
    for (int r = 0; r < 8; ++r) pd[r] = 0.0f;
#pragma unroll
    for (int n = 0; n < 13; ++n) {
      int e = n * 16 + l15;
      float bv = (e < EE) ? b_att[k * EE + e] : 0.0f;
      float qv = (e < EE) ? q_att[k * EE + e] : 0.0f;
#pragma unroll
      for (int r = 0; r < 8; ++r) pd[r] += tanh_fast(acc[n][r] + bv) * qv;
    }
#pragma unroll
    for (int r = 0; r < 8; ++r) {
      pd[r] += __shfl_xor(pd[r], 1, 32);
      pd[r] += __shfl_xor(pd[r], 2, 32);
      pd[r] += __shfl_xor(pd[r], 4, 32);
      pd[r] += __shfl_xor(pd[r], 8, 32);
    }
    if (l15 == 0)
#pragma unroll
      for (int r = 0; r < 8; ++r) logits[wave * 16 + khalf + r] = pd[r];
  }
  __syncthreads();

  /* softmax over h<100 (one wave) */
  if (wave == 0) {
    float mx = -1e30f;
    for (int h = lane; h < HH; h += 32) mx = fmaxf(mx, logits[h]);
    for (int off = 16; off > 0; off >>= 1) mx = fmaxf(mx, __shfl_xor(mx, off, 32));
    float s = 0.0f;
    for (int h = lane; h < HH; h += 32) { float ev = expf(logits[h] - mx); aw[h] = ev; s += ev; }
    for (int off = 16; off > 0; off >>= 1) s += __shfl_xor(s, off, 32);
    float inv = 1.0f / s;
    for (int h = lane; h < HH; h += 32) aw[h] *= inv;
  }
  __syncthreads();

  /* u[d] = sum_h a[h] * his[h][d] */
  for (int d = tid; d < DD; d += 256) {
    float s = 0.0f;
    for (int h = 0; h < HH; ++h) s += aw[h] * (float)sA[h * SA_STR + d];
    u_s[d] = s;
  }
  __syncthreads();

  /* write u_channel = broadcast over C (output region after scores) */
  float* out_u = out + NTOT;
  for (int c = 0; c < CC; ++c) {
    size_t baseo = (((size_t)(b * CC + c)) * KCH + k) * DD;
    for (int d = tid; d < DD; d += 256) out_u[baseo + d] = u_s[d];
  }
  /* g1[b,k,e] = b_lin[e] + sum_d u[d] * W_lin[d,e] */
  if (tid < EE) {
    int e = tid;
    float s = b_lin[e];
    for (int d = 0; d < DD; ++d) s += u_s[d] * W_lin[(size_t)d * EE + e];
    g1ws[(b * KCH + k) * EE + e] = s;
  }
}

/* =====================================================================
 * Kernel 2: 64 (n,k) rows per block (16 candidates). 4 waves, one M-tile each.
 * ===================================================================== */
__global__ void __launch_bounds__(128, 1) comb_kernel(
    const float* __restrict__ cdd,   const float* __restrict__ chw,
    const float* __restrict__ W_lin, const float* __restrict__ q_vec,
    const float* __restrict__ g1ws,  float* __restrict__ out)
{
  extern __shared__ char smem[];
  __bf16* sA  = (__bf16*)smem;                                  /* [64][136]  */
  __bf16* sBt = (__bf16*)(smem + K2_ROWS * SA2_STR * 2);        /* [208][136] */
  float* logits  = (float*)(smem + K2_ROWS * SA2_STR * 2 + SBT_COLS * SB2_STR * 2);
  float* score_s = logits + K2_ROWS;

  const int blk = blockIdx.x;
  const int tid = threadIdx.x;
  const int wave = tid >> 5, lane = tid & 31;
  const int l15 = lane & 15;
  const int khalf = (lane < 16) ? 0 : 8;
  const int kb2   = (lane < 16) ? 0 : 16;
  const int row0 = blk * K2_ROWS;

  /* stage A: channel_weight rows (pad 100 -> 128 cols) */
  for (int idx = tid; idx < K2_ROWS * 128; idx += 128) {
    int lr = idx >> 7, f = idx & 127;
    float v = (f < CWF) ? chw[(size_t)(row0 + lr) * CWF + f] : 0.0f;
    sA[lr * SA2_STR + f] = (__bf16)v;
  }
  /* stage B^T: W_lin[400+f][e] */
  for (int idx = tid; idx < 128 * SBT_COLS; idx += 128) {
    int f = idx / SBT_COLS, e = idx - f * SBT_COLS;
    float v = (f < CWF && e < EE) ? W_lin[(size_t)(DD + f) * EE + e] : 0.0f;
    sBt[e * SB2_STR + f] = (__bf16)v;
  }
  __syncthreads();

  v16bf afrag[4];
#pragma unroll
  for (int kc = 0; kc < 4; ++kc)
    afrag[kc] = load_a_frag(sA, SA2_STR, wave * 16 + l15, kc * 32 + khalf);

  float pd[8];
#pragma unroll
  for (int r = 0; r < 8; ++r) pd[r] = 0.0f;

  for (int n = 0; n < 13; ++n) {
    v8f acc;
#pragma unroll
    for (int r = 0; r < 8; ++r) acc[r] = 0.0f;
#pragma unroll
    for (int kc = 0; kc < 4; ++kc) {
      v16bf bfrag = load_b_frag(sBt, SB2_STR, n * 16 + l15, kc * 32 + kb2);
      acc = __builtin_amdgcn_wmma_f32_16x16x32_bf16(
          false, afrag[kc], false, bfrag, (short)0, acc, false, false);
    }
    int e = n * 16 + l15;
    float qv = (e < EE) ? q_vec[e] : 0.0f;
#pragma unroll
    for (int r = 0; r < 8; ++r) {
      int g = row0 + wave * 16 + khalf + r;        /* global (n,k) row */
      float gv = (e < EE) ? g1ws[((g >> 7) * KCH + (g & 3)) * EE + e] : 0.0f;
      pd[r] += tanh_fast(acc[r] + gv) * qv;
    }
  }
#pragma unroll
  for (int r = 0; r < 8; ++r) {
    pd[r] += __shfl_xor(pd[r], 1, 32);
    pd[r] += __shfl_xor(pd[r], 2, 32);
    pd[r] += __shfl_xor(pd[r], 4, 32);
    pd[r] += __shfl_xor(pd[r], 8, 32);
  }
  if (l15 == 0)
#pragma unroll
    for (int r = 0; r < 8; ++r) logits[wave * 16 + khalf + r] = pd[r];

  /* score[row] = dot(u_channel[row,:], cdd[row,:]) — coalesced per-wave */
  const float* out_u = out + NTOT;
  for (int lr = wave * 16; lr < wave * 16 + 16; ++lr) {
    size_t base = (size_t)(row0 + lr) * DD;
    float s = 0.0f;
    for (int d = lane; d < DD; d += 32) s += out_u[base + d] * cdd[base + d];
    for (int off = 16; off > 0; off >>= 1) s += __shfl_xor(s, off, 32);
    if (lane == 0) score_s[lr] = s;
  }
  __syncthreads();

  /* softmax over K=4 and combine */
  if (tid < K2_NPB) {
    int base = tid * KCH;
    float mx = -1e30f;
#pragma unroll
    for (int k = 0; k < KCH; ++k) mx = fmaxf(mx, logits[base + k]);
    float s = 0.0f, w[KCH];
#pragma unroll
    for (int k = 0; k < KCH; ++k) { w[k] = expf(logits[base + k] - mx); s += w[k]; }
    float sc = 0.0f;
#pragma unroll
    for (int k = 0; k < KCH; ++k) sc += (w[k] / s) * score_s[base + k];
    out[blk * K2_NPB + tid] = sc;
  }
}

extern "C" void kernel_launch(void* const* d_in, const int* in_sizes, int n_in,
                              void* d_out, int out_size, void* d_ws, size_t ws_size,
                              hipStream_t stream) {
  const float* his   = (const float*)d_in[0];
  const float* cdd   = (const float*)d_in[1];
  const float* chw   = (const float*)d_in[2];
  const float* W_att = (const float*)d_in[3];
  const float* b_att = (const float*)d_in[4];
  const float* q_att = (const float*)d_in[5];
  const float* W_lin = (const float*)d_in[6];
  const float* b_lin = (const float*)d_in[7];
  const float* q_vec = (const float*)d_in[8];
  float* out  = (float*)d_out;
  float* g1ws = (float*)d_ws;     /* B*K*E = 409,600 bytes */

  const size_t shm1 = (size_t)SA_ROWS * SA_STR * 2 + (size_t)SBT_COLS * SBT_STR * 2
                    + (2 * SA_ROWS + DD) * sizeof(float);            /* ~114 KB */
  const size_t shm2 = (size_t)K2_ROWS * SA2_STR * 2 + (size_t)SBT_COLS * SB2_STR * 2
                    + (2 * K2_ROWS) * sizeof(float);                 /* ~74 KB  */

  attn_kernel<<<dim3(BB * KCH), dim3(256), shm1, stream>>>(
      his, W_att, b_att, q_att, W_lin, b_lin, out, g1ws);
  comb_kernel<<<dim3(NTOT / K2_NPB), dim3(128), shm2, stream>>>(
      cdd, chw, W_lin, q_vec, g1ws, out);
}
